// MultiHeadAttention_42520176230572
// MI455X (gfx1250) — matrix operand order repaired and checked by reading, not compiled, not run on hardware
//
#include <hip/hip_runtime.h>
#include <hip/hip_bf16.h>

#define DEV __device__ __forceinline__

typedef __attribute__((ext_vector_type(16))) __bf16 v16bf;
typedef __attribute__((ext_vector_type(8)))  float  v8f;
typedef __attribute__((ext_vector_type(4)))  unsigned int u32x4;
typedef __attribute__((ext_vector_type(8)))  int          i32x8;
typedef __attribute__((ext_vector_type(4)))  int          i32x4;

union Frag { v16bf v; unsigned u[8]; uint4 q[2]; };

static constexpr int Bz = 2, S = 2048, D = 1024, H = 16, HD = 64;
static constexpr int LD = 2052;   // padded LDS row stride (floats); 2052 % 64 = 4

// round-to-nearest-even fp32 -> bf16 (integer ops only)
DEV unsigned short f2bf_bits(float f) {
  unsigned u = __float_as_uint(f);
  u += 0x7FFFu + ((u >> 16) & 1u);
  return (unsigned short)(u >> 16);
}
DEV unsigned pack_bf16x2(float lo, float hi) {
  unsigned ul = __float_as_uint(lo); ul += 0x7FFFu + ((ul >> 16) & 1u);
  unsigned uh = __float_as_uint(hi); uh += 0x7FFFu + ((uh >> 16) & 1u);
  return (ul >> 16) | (uh & 0xFFFF0000u);
}
DEV v8f wmma_bf16(v16bf a, v16bf b, v8f c) {
  return __builtin_amdgcn_wmma_f32_16x16x32_bf16(false, a, false, b, (short)0, c, false, false);
}

// TDM: async-load a [rows x 64] bf16 tile (row stride 64 elems) global -> LDS.
// D# per CDNA5 ISA ch.8: group0 {count, lds_addr, global_addr, type=2},
// group1 {data_size=2B, tensor_dim0=64, tensor_dim1=rows, tile_dim0=64,
//         tile_dim1=rows, tensor_dim0_stride=64}; groups 2/3 zero (2D tensor).
// This toolchain exposes the 6-arg builtin form (extra i32x8 group, zero).
DEV void tdm_load_tile_64xN(const void* gptr, unsigned lds_byte_addr, unsigned rows) {
#if __has_builtin(__builtin_amdgcn_tensor_load_to_lds)
  const unsigned long long ga = (unsigned long long)gptr;
  u32x4 g0;
  g0[0] = 1u;                                            // count = 1 valid descriptor
  g0[1] = lds_byte_addr;                                 // lds_addr
  g0[2] = (unsigned)ga;                                  // global_addr[31:0]
  g0[3] = (unsigned)((ga >> 32) & 0x01FFFFFFu) | (2u << 30);  // addr[56:32] | type=2
  i32x8 g1;
  g1[0] = (int)(1u << 16);                               // data_size = 2 bytes
  g1[1] = (int)(64u << 16);                              // tensor_dim0 = 64
  g1[2] = (int)(rows << 16);                             // tensor_dim1 = rows
  g1[3] = (int)(64u << 16);                              // tile_dim0 = 64
  g1[4] = (int)rows;                                     // tile_dim1 = rows
  g1[5] = 64;                                            // tensor_dim0_stride = 64
  g1[6] = 0; g1[7] = 0;
  i32x4 g2 = {0, 0, 0, 0}, g3 = {0, 0, 0, 0};
  i32x8 g4 = {0, 0, 0, 0, 0, 0, 0, 0};
  __builtin_amdgcn_tensor_load_to_lds(g0, g1, g2, g3, g4, 0);
#else
  (void)gptr; (void)lds_byte_addr; (void)rows;
#endif
}

// ---------------------------------------------------------------------------
// Weight transpose + fp32->bf16:  W[k][n] -> Wt[n][k] bf16 (K-contiguous cols)
// ---------------------------------------------------------------------------
__global__ void transpose_to_bf16(const float* __restrict__ W, unsigned short* __restrict__ Wt) {
  __shared__ float tile[32][33];
  const int bx = blockIdx.x * 32, by = blockIdx.y * 32;
  const int tx = threadIdx.x, ty = threadIdx.y;
#pragma unroll
  for (int i = 0; i < 32; i += 8)
    tile[ty + i][tx] = W[(size_t)(by + ty + i) * D + bx + tx];
  __syncthreads();
#pragma unroll
  for (int i = 0; i < 32; i += 8)
    Wt[(size_t)(bx + ty + i) * D + by + tx] = f2bf_bits(tile[tx][ty + i]);
}

// ---------------------------------------------------------------------------
// GEMM: out = X[4096 x 1024] @ Wt^T + bias.  bf16 WMMA, fp32 accum.
// Wave = 16x64 tile; 2-stage register pipeline on the K loop.
// MODE 0/1 -> [B,H,S,HD] bf16; MODE 2 -> [B,H,HD,S] bf16; MODE 3 -> fp32 final.
// ---------------------------------------------------------------------------
template <int MODE, bool SRCBF>
__global__ void proj_kernel(const void* __restrict__ Xv,
                            const unsigned short* __restrict__ Wt,
                            const float* __restrict__ bias,
                            void* __restrict__ outv) {
  const int lane = threadIdx.x & 31;
  const int wave = blockIdx.x * (blockDim.x >> 5) + (threadIdx.x >> 5);
  const int m0 = (wave >> 4) * 16, n0 = (wave & 15) * 64;
  const int lrow = lane & 15, hi = lane >> 4;
  const int arow = m0 + lrow;

  const float*          Xf = (const float*)Xv;
  const unsigned short* Xb = (const unsigned short*)Xv;

  auto loadA = [&](int kc, Frag& a) {
    const int k0 = kc * 32;
    if constexpr (SRCBF) {
      const uint4* xr = (const uint4*)(Xb + (size_t)arow * D);
      a.q[0] = xr[(k0 + hi * 8) >> 3];
      a.q[1] = xr[(k0 + hi * 8 + 16) >> 3];
    } else {
      // A layout (16-bit 16x32): lanes 0-15: K k0+0..7 & k0+16..23; lanes 16-31: +8
      const float* xr = Xf + (size_t)arow * D + k0 + hi * 8;
      float4 f0 = ((const float4*)xr)[0];
      float4 f1 = ((const float4*)xr)[1];
      float4 f2 = ((const float4*)(xr + 16))[0];
      float4 f3 = ((const float4*)(xr + 16))[1];
      a.u[0] = pack_bf16x2(f0.x, f0.y); a.u[1] = pack_bf16x2(f0.z, f0.w);
      a.u[2] = pack_bf16x2(f1.x, f1.y); a.u[3] = pack_bf16x2(f1.z, f1.w);
      a.u[4] = pack_bf16x2(f2.x, f2.y); a.u[5] = pack_bf16x2(f2.z, f2.w);
      a.u[6] = pack_bf16x2(f3.x, f3.y); a.u[7] = pack_bf16x2(f3.z, f3.w);
    }
  };
  auto loadB = [&](int kc, Frag* bb) {
    const int k0 = kc * 32;
#pragma unroll
    for (int j = 0; j < 4; ++j) {
      const int n = n0 + j * 16 + lrow;
      const uint4* wr = (const uint4*)(Wt + (size_t)n * D + k0 + hi * 8);
      bb[j].q[0] = wr[0]; bb[j].q[1] = wr[2];
    }
  };

  v8f acc[4] = {};
  Frag a[2], bm[2][4];
  loadA(0, a[0]); loadB(0, bm[0]);
  for (int kc = 0; kc < D / 32; ++kc) {
    const int cur = kc & 1;
    if (kc + 1 < D / 32) { loadA(kc + 1, a[cur ^ 1]); loadB(kc + 1, bm[cur ^ 1]); }
#pragma unroll
    for (int j = 0; j < 4; ++j) acc[j] = wmma_bf16(a[cur].v, bm[cur][j].v, acc[j]);
  }

#pragma unroll
  for (int j = 0; j < 4; ++j) {
    const int n = n0 + j * 16 + lrow;
    const float bn = bias[n];
#pragma unroll
    for (int r = 0; r < 8; ++r) {
      const int m = m0 + r + hi * 8;
      const float val = acc[j][r] + bn;
      if constexpr (MODE == 3) {
        ((float*)outv)[(size_t)m * D + n] = val;
      } else {
        const int bb = m / S, s = m % S, hh = n / HD, hd = n % HD;
        unsigned short* o = (unsigned short*)outv;
        size_t idx;
        if constexpr (MODE == 2) idx = (((size_t)(bb * H + hh)) * HD + hd) * S + s;
        else                     idx = (((size_t)(bb * H + hh)) * S + s) * HD + hd;
        o[idx] = f2bf_bits(val);
      }
    }
  }
}

// ---------------------------------------------------------------------------
// Fused attention. WG = 4 waves, 16 query rows, ~148 KB LDS -> 2 WGs/WGP.
//  scores: split-K, each wave 512 keys; K tiles staged via TDM, 2 deep.
//  softmax: 4 rows/wave, b128 LDS + b128 global attn write.
//  P@V: split-N, each wave one 16-wide HD tile over all 2048 keys.
// ---------------------------------------------------------------------------
__global__ void attn_kernel(const unsigned short* __restrict__ Qh,
                            const unsigned short* __restrict__ Kh,
                            const unsigned short* __restrict__ Vt,
                            unsigned short* __restrict__ ctx,
                            float* __restrict__ attn_out) {
  extern __shared__ float smem[];
  const int wave = threadIdx.x >> 5, lane = threadIdx.x & 31;
  const int lrow = lane & 15, hi = lane >> 4;
  const int b = blockIdx.z, h = blockIdx.y;
  const int q0 = blockIdx.x * 16;

  const unsigned short* Qbh = Qh + ((size_t)(b * H + h)) * S * HD;   // [S][HD]
  const unsigned short* Kbh = Kh + ((size_t)(b * H + h)) * S * HD;   // [S][HD]
  const unsigned short* Vbh = Vt + ((size_t)(b * H + h)) * HD * S;   // [HD][S]

  float* scores = smem;                              // [16][LD]
  float* invs   = smem + 16 * LD;                    // [16]
  float* kstg   = smem + 16 * LD + 16 + wave * 1024; // per-wave 2 x 2KB K tiles
  const unsigned kstg_addr = (unsigned)(size_t)kstg; // flat->LDS: low 32 bits

  // Q A-fragments (HD=64 -> two K=32 chunks), loaded once per wave
  Frag aq[2];
  {
    const uint4* qr = (const uint4*)(Qbh + (size_t)(q0 + lrow) * HD);
#pragma unroll
    for (int kc = 0; kc < 2; ++kc) {
      aq[kc].q[0] = qr[(kc * 32 + hi * 8) >> 3];
      aq[kc].q[1] = qr[(kc * 32 + hi * 8 + 16) >> 3];
    }
  }

  // ---- scores = (Q K^T)/8 : 32 K-tiles of 16 keys, TDM double-buffered -----
  const int krow0 = wave * 512;
  tdm_load_tile_64xN(Kbh + (size_t)krow0 * HD, kstg_addr, 16);
  for (int kt = 0; kt < 32; ++kt) {
    if (kt + 1 < 32) {
      tdm_load_tile_64xN(Kbh + (size_t)(krow0 + (kt + 1) * 16) * HD,
                         kstg_addr + ((kt + 1) & 1) * 2048, 16);
      __builtin_amdgcn_s_wait_tensorcnt(1);
    } else {
      __builtin_amdgcn_s_wait_tensorcnt(0);
    }
    const uint4* tk = (const uint4*)(kstg + (kt & 1) * 512);   // staged [16][64] bf16
    v8f acc = {};
#pragma unroll
    for (int kc = 0; kc < 2; ++kc) {
      Frag bk;                                  // lane lrow = key row, K in table order
      const int idx = lrow * 8 + kc * 4 + hi;                   // uint4 units
      bk.q[0] = tk[idx]; bk.q[1] = tk[idx + 2];
      acc = wmma_bf16(aq[kc].v, bk.v, acc);
    }
    const int col = krow0 + kt * 16 + lrow;
#pragma unroll
    for (int r = 0; r < 8; ++r) scores[(r + hi * 8) * LD + col] = acc[r] * 0.125f;
  }
  __syncthreads();

  // ---- softmax: 4 rows per wave, vectorized b128 passes --------------------
  for (int rr = 0; rr < 4; ++rr) {
    const int r = wave * 4 + rr;
    float4* row4 = (float4*)(scores + r * LD);
    float mx = -3.402823466e38f;
    for (int i = lane; i < S / 4; i += 32) {
      float4 v = row4[i];
      mx = fmaxf(mx, fmaxf(fmaxf(v.x, v.y), fmaxf(v.z, v.w)));
    }
#pragma unroll
    for (int off = 16; off; off >>= 1) mx = fmaxf(mx, __shfl_xor(mx, off, 32));
    float sum = 0.f;
    for (int i = lane; i < S / 4; i += 32) {
      float4 v = row4[i];
      v.x = __expf(v.x - mx); v.y = __expf(v.y - mx);
      v.z = __expf(v.z - mx); v.w = __expf(v.w - mx);
      row4[i] = v;
      sum += (v.x + v.y) + (v.z + v.w);
    }
#pragma unroll
    for (int off = 16; off; off >>= 1) sum += __shfl_xor(sum, off, 32);
    const float is = 1.0f / sum;
    if (lane == 0) invs[r] = is;
    float4* ao4 = (float4*)(attn_out + (((size_t)(b * H + h)) * S + q0 + r) * S);
    for (int i = lane; i < S / 4; i += 32) {
      float4 v = row4[i];
      v.x *= is; v.y *= is; v.z *= is; v.w *= is;
      ao4[i] = v;                                        // b128, coalesced
    }
  }
  __syncthreads();

  // ---- out = (E @ V) * inv_sum : wave owns HD cols [wave*16, wave*16+16) ---
  v8f oacc = {};
  const int n = wave * 16 + lrow;
  const float4* er4 = (const float4*)(scores + lrow * LD);
  const uint4* vr = (const uint4*)(Vbh + (size_t)n * S);
  for (int kt = 0; kt < 64; ++kt) {
    if (kt + 1 < 64) __builtin_prefetch(vr + (kt + 1) * 4, 0, 0);
    Frag bv;
    bv.q[0] = vr[kt * 4 + hi];
    bv.q[1] = vr[kt * 4 + hi + 2];
    Frag ea;                                   // A from LDS E, b128 LDS loads
    const int kb = kt * 32 + hi * 8;
    float4 f0 = er4[kb >> 2],        f1 = er4[(kb >> 2) + 1];
    float4 f2 = er4[(kb + 16) >> 2], f3 = er4[((kb + 16) >> 2) + 1];
    ea.u[0] = pack_bf16x2(f0.x, f0.y); ea.u[1] = pack_bf16x2(f0.z, f0.w);
    ea.u[2] = pack_bf16x2(f1.x, f1.y); ea.u[3] = pack_bf16x2(f1.z, f1.w);
    ea.u[4] = pack_bf16x2(f2.x, f2.y); ea.u[5] = pack_bf16x2(f2.z, f2.w);
    ea.u[6] = pack_bf16x2(f3.x, f3.y); ea.u[7] = pack_bf16x2(f3.z, f3.w);
    oacc = wmma_bf16(ea.v, bv.v, oacc);
  }
#pragma unroll
  for (int r = 0; r < 8; ++r) {
    const int row = r + hi * 8;
    const float val = oacc[r] * invs[row];
    ctx[((size_t)(b * S + q0 + row)) * D + h * HD + n] = f2bf_bits(val);
  }
}

// ---------------------------------------------------------------------------
extern "C" void kernel_launch(void* const* d_in, const int* in_sizes, int n_in,
                              void* d_out, int out_size, void* d_ws, size_t ws_size,
                              hipStream_t stream) {
  (void)in_sizes; (void)n_in; (void)out_size; (void)ws_size;

  const float* query = (const float*)d_in[0];
  const float* key_  = (const float*)d_in[1];
  const float* value = (const float*)d_in[2];
  const float* W_q = (const float*)d_in[3];  const float* b_q = (const float*)d_in[4];
  const float* W_k = (const float*)d_in[5];  const float* b_k = (const float*)d_in[6];
  const float* W_v = (const float*)d_in[7];  const float* b_v = (const float*)d_in[8];
  const float* W_o = (const float*)d_in[9];  const float* b_o = (const float*)d_in[10];

  char* ws = (char*)d_ws;
  const size_t WB = (size_t)D * D * sizeof(unsigned short);        // 2 MB / weight
  const size_t AB = (size_t)Bz * S * D * sizeof(unsigned short);   // 8 MB / activation
  unsigned short* Wq_t = (unsigned short*)(ws + 0 * WB);
  unsigned short* Wk_t = (unsigned short*)(ws + 1 * WB);
  unsigned short* Wv_t = (unsigned short*)(ws + 2 * WB);
  unsigned short* Wo_t = (unsigned short*)(ws + 3 * WB);
  unsigned short* Qh   = (unsigned short*)(ws + 4 * WB);
  unsigned short* Kh   = (unsigned short*)(ws + 4 * WB + 1 * AB);
  unsigned short* Vt   = (unsigned short*)(ws + 4 * WB + 2 * AB);
  unsigned short* ctxp = (unsigned short*)(ws + 4 * WB + 3 * AB);

  float* final_out = (float*)d_out;
  float* attn_out  = final_out + (size_t)Bz * S * D;   // outputs: final, then attn

  dim3 tb(32, 8), tg(D / 32, D / 32);
  transpose_to_bf16<<<tg, tb, 0, stream>>>(W_q, Wq_t);
  transpose_to_bf16<<<tg, tb, 0, stream>>>(W_k, Wk_t);
  transpose_to_bf16<<<tg, tb, 0, stream>>>(W_v, Wv_t);
  transpose_to_bf16<<<tg, tb, 0, stream>>>(W_o, Wo_t);

  proj_kernel<0, false><<<1024, 128, 0, stream>>>(query, Wq_t, b_q, Qh);
  proj_kernel<1, false><<<1024, 128, 0, stream>>>(key_,  Wk_t, b_k, Kh);
  proj_kernel<2, false><<<1024, 128, 0, stream>>>(value, Wv_t, b_v, Vt);

  // LDS: 16*LD scores + 16 invs + 4 waves * 2 * 2KB K staging  (~148 KB)
  const size_t shbytes = (size_t)(16 * LD + 16 + 4 * 1024) * sizeof(float);
  attn_kernel<<<dim3(S / 16, H, Bz), 128, shbytes, stream>>>(Qh, Kh, Vt, ctxp, attn_out);

  proj_kernel<3, true><<<1024, 128, 0, stream>>>(ctxp, Wo_t, b_o, final_out);
}
